// PNet_79396765434068
// MI455X (gfx1250) — matrix operand order repaired and checked by reading, not compiled, not run on hardware
//
#include <hip/hip_runtime.h>
#include <hip/hip_bf16.h>

#define EPSC 1e-5f

typedef __attribute__((ext_vector_type(16))) _Float16 v16h;
typedef __attribute__((ext_vector_type(8)))  _Float16 v8h;
typedef __attribute__((ext_vector_type(8)))  float    v8f;

// ---------------------------------------------------------------------------
// Weight packing: f32 source -> f16 (taps, CoutP, CinP), zero padded.
// mode 0: conv weights  (cout, cin, kt, 1):   w[(m*cin+k)*kt + tap]
// mode 1: deconv weights (cin, cout, kt, 1), flip+transpose:
//         eff[m][k][tap] = w[(k*cout+m)*kt + (kt-1-tap)]
// ---------------------------------------------------------------------------
__global__ void pack_w_kernel(const float* __restrict__ w, _Float16* __restrict__ wp,
                              int cout, int cin, int kt, int coutP, int cinP, int mode) {
    int idx = blockIdx.x * 256 + threadIdx.x;
    int tot = kt * coutP * cinP;
    if (idx >= tot) return;
    int k   = idx % cinP;
    int m   = (idx / cinP) % coutP;
    int tap = idx / (cinP * coutP);
    float val = 0.f;
    if (m < cout && k < cin) {
        if (mode == 0) val = w[(m * cin + k) * kt + tap];
        else           val = w[(k * cout + m) * kt + (kt - 1 - tap)];
    }
    wp[idx] = (_Float16)val;
}

// obs_pose (N,P,T,V,1) + data_bn -> x (n,t,v, c<CP) f16
__global__ void prep_pose_kernel(const float* __restrict__ pose,
                                 const float* __restrict__ g,
                                 const float* __restrict__ b,
                                 _Float16* __restrict__ x,
                                 int N, int T, int V, int P, int CP) {
    int idx = blockIdx.x * 256 + threadIdx.x;
    int tot = N * T * V * CP;
    if (idx >= tot) return;
    int c = idx % CP;
    int r = idx / CP;
    int v = r % V;
    int t = (r / V) % T;
    int n = r / (V * T);
    float val = 0.f;
    if (c < P) {
        int ch = v * P + c;                       // data_bn channel order (V,P)
        float s = g[ch] * rsqrtf(1.f + EPSC);
        val = pose[((n * P + c) * T + t) * V + v] * s + b[ch];
    }
    x[idx] = (_Float16)val;
}

// f32 (col, cin) -> f16 (col, CP) zero padded  (loc / gridflow)
__global__ void pack_cols_kernel(const float* __restrict__ src, _Float16* __restrict__ dst,
                                 int cols, int cin, int CP) {
    int idx = blockIdx.x * 256 + threadIdx.x;
    int tot = cols * CP;
    if (idx >= tot) return;
    int c = idx % CP;
    int col = idx / CP;
    dst[idx] = (c < cin) ? (_Float16)src[(long)col * cin + c] : (_Float16)0.f;
}

// rec output (n,t,v, p<srcCP) -> pose_in (n,t, c=v*P+p <dstCP)
__global__ void pose_in_kernel(const _Float16* __restrict__ src, _Float16* __restrict__ dst,
                               int N, int T, int V, int P, int srcCP, int dstCP) {
    int idx = blockIdx.x * 256 + threadIdx.x;
    int tot = N * T * dstCP;
    if (idx >= tot) return;
    int c = idx % dstCP;
    long nt = idx / dstCP;
    _Float16 val = (_Float16)0.f;
    if (c < V * P) {
        int v = c / P, p = c % P;
        val = src[(nt * V + v) * srcCP + p];
    }
    dst[idx] = val;
}

// concat three (col,128) buffers into (col,384)
__global__ void concat3_kernel(const _Float16* __restrict__ a, const _Float16* __restrict__ b,
                               const _Float16* __restrict__ c, _Float16* __restrict__ dst,
                               int cols) {
    int idx = blockIdx.x * 256 + threadIdx.x;
    int tot = cols * 384;
    if (idx >= tot) return;
    int ch = idx % 384;
    int col = idx / 384;
    _Float16 v;
    if (ch < 128)      v = a[(long)col * 128 + ch];
    else if (ch < 256) v = b[(long)col * 128 + (ch - 128)];
    else               v = c[(long)col * 128 + (ch - 256)];
    dst[idx] = v;
}

// ---------------------------------------------------------------------------
// Graph einsum:  z[n,t,w,c] = relu(bn1( sum_{k,v} y[n,t,v, k*cout+c] * (A*imp)[k,v,w] ))
// one block per (n,t); A*imp and the y tile staged in LDS.
// ---------------------------------------------------------------------------
__global__ void gcn_einsum_kernel(const _Float16* __restrict__ Yg,
                                  const float* __restrict__ A,
                                  const float* __restrict__ imp,
                                  const float* __restrict__ g1,
                                  const float* __restrict__ b1,
                                  _Float16* __restrict__ Z,
                                  int N, int T, int V, int cout, int KCP, int CoutP) {
    __shared__ float   Ai[3 * 18 * 18];
    __shared__ _Float16 yt[18 * 768];
    long nt = blockIdx.x;
    for (int i = threadIdx.x; i < 3 * V * V; i += blockDim.x) Ai[i] = A[i] * imp[i];
    const _Float16* src = Yg + nt * V * KCP;
    for (int i = threadIdx.x; i < V * KCP; i += blockDim.x) yt[i] = src[i];
    __syncthreads();
    int tot = V * CoutP;
    for (int o = threadIdx.x; o < tot; o += blockDim.x) {
        int c = o % CoutP;
        int w = o / CoutP;
        float val = 0.f;
        if (c < cout) {
            float acc = 0.f;
            for (int k = 0; k < 3; ++k)
                for (int v = 0; v < V; ++v)
                    acc += (float)yt[v * KCP + k * cout + c] * Ai[(k * V + v) * V + w];
            float s = g1[c] * rsqrtf(1.f + EPSC);
            val = fmaxf(acc * s + b1[c], 0.f);
        }
        Z[(nt * V + w) * CoutP + c] = (_Float16)val;
    }
}

// ---------------------------------------------------------------------------
// Generic WMMA GEMM, m-register-blocked: one wave computes a (MBLK*16) x 16
// output tile. Per k-block: ONE activation (B) fragment load shared by MBLK
// weight (A) fragments / WMMAs -> MBLK x less L2 activation traffic and MBLK
// back-to-back v_wmma ops per load (hides XDL latency).
// f16 inputs, fp32 accumulate via v_wmma_f32_16x16x32_f16.
// Epilogue: (+bias)*bnScale(+beta)(+res)(relu), optional fp32 store.
//
// Fragment layouts used by this kernel (wave32):
//   A 16x32 f16: lane m = lane&15; half j -> K = j + (lane>=16?16:0)
//   B 32x16 f16: lane n = lane&15; half j -> K = j + (lane>=16?16:0)
//   C/D 16x16 f32: vgpr r -> M = r + (lane>=16?8:0), N = lane&15
// ---------------------------------------------------------------------------
template <int MBLK>
__global__ void __launch_bounds__(256)
gemm_wmma_kernel(const _Float16* __restrict__ Wp,
                 const _Float16* __restrict__ X,
                 _Float16* __restrict__ Y,
                 const _Float16* __restrict__ Res,
                 const float* __restrict__ bias,
                 const float* __restrict__ gamma,
                 const float* __restrict__ beta,
                 float* __restrict__ out32, int o32s,
                 int Nb, int Tin, int Tout, int V,
                 int CinP, int CoutP, int cout,
                 int taps, int pad, int relu) {
    const int lane = threadIdx.x & 31;
    const int wave = threadIdx.x >> 5;
    const int ncols = Nb * Tout * V;
    const int ncolT = (ncols + 15) >> 4;
    const int nmG   = (CoutP >> 4) / MBLK;          // m-tile groups
    const long tile = (long)blockIdx.x * (blockDim.x >> 5) + wave;
    if (tile >= (long)ncolT * nmG) return;          // wave-uniform exit
    const int mg = (int)(tile / ncolT);
    const int cb = (int)(tile % ncolT);
    const int lhalf = lane >> 4;                    // 0 | 1
    const int lmod  = lane & 15;
    const int col = cb * 16 + lmod;
    const int TV = Tout * V;
    const int n = col / TV;
    const int rem = col % TV;
    const int t = rem / V;
    const int v = rem % V;
    const bool colOK = col < ncols;

    v8f acc[MBLK];
#pragma unroll
    for (int i = 0; i < MBLK; ++i) acc[i] = (v8f){0.f, 0.f, 0.f, 0.f, 0.f, 0.f, 0.f, 0.f};
    const v16h bzero = {};
    // A-fragment base for m-subtile 0 of this group
    const _Float16* wbase =
        Wp + ((long)(mg * MBLK * 16 + lmod)) * CinP + lhalf * 16;

    for (int tap = 0; tap < taps; ++tap) {
        const int tin = t + tap - pad;
        const bool ok = colOK && (tin >= 0) && (tin < Tin);
        const int tc = tin < 0 ? 0 : (tin >= Tin ? Tin - 1 : tin);
        const _Float16* xb = X + ((long)((n * Tin + tc) * V + v)) * CinP + lhalf * 16;
        const _Float16* wb = wbase + (long)tap * CoutP * CinP;
        for (int kb = 0; kb < CinP; kb += 32) {
            v16h bfrag = ok ? *(const v16h*)(xb + kb) : bzero;
#pragma unroll
            for (int i = 0; i < MBLK; ++i) {
                const _Float16* wi = wb + (long)i * 16 * CinP + kb;
                v8h a0 = *(const v8h*)(wi);
                v8h a1 = *(const v8h*)(wi + 8);
                v16h afrag = __builtin_shufflevector(a0, a1,
                    0, 1, 2, 3, 4, 5, 6, 7, 8, 9, 10, 11, 12, 13, 14, 15);
                acc[i] = __builtin_amdgcn_wmma_f32_16x16x32_f16(
                    false, afrag, false, bfrag, (short)0, acc[i], false, false);
            }
        }
    }

    const long ybase = (long)col * CoutP;
#pragma unroll
    for (int i = 0; i < MBLK; ++i) {
        const int mbase = (mg * MBLK + i) * 16 + lhalf * 8;
#pragma unroll
        for (int r = 0; r < 8; ++r) {
            int ch = mbase + r;
            float val = 0.f;
            if (ch < cout) {
                val = acc[i][r];
                if (bias)  val += bias[ch];
                if (gamma) val *= gamma[ch] * rsqrtf(1.f + EPSC);
                if (beta)  val += beta[ch];
                if (Res)   val += (float)Res[ybase + ch];
                if (relu)  val = fmaxf(val, 0.f);
            }
            if (colOK) {
                Y[ybase + ch] = (_Float16)val;
                if (out32 && ch < o32s) out32[(long)col * o32s + ch] = val;
            }
        }
    }
}

// ---------------------------------------------------------------------------
// Host orchestration
// ---------------------------------------------------------------------------
struct STIdx { int b1, b2, g1, g2, gb, gw, imp, rb, rbb, rg, rw, tb, tw; };

extern "C" void kernel_launch(void* const* d_in, const int* in_sizes, int n_in,
                              void* d_out, int out_size, void* d_ws, size_t ws_size,
                              hipStream_t stream) {
    (void)in_sizes; (void)n_in; (void)out_size; (void)ws_size;
    const int N = 512, T = 10, V = 18, P = 3, K = 3, L = 2, G = 24;
    auto F = [&](int i) { return (const float*)d_in[i]; };
    auto rup32 = [](int x) { return (x + 31) & ~31; };

    size_t cur = 0;
    auto alloc16 = [&](size_t elems) -> _Float16* {
        _Float16* p = (_Float16*)((char*)d_ws + cur);
        cur = (cur + elems * sizeof(_Float16) + 255) & ~(size_t)255;
        return p;
    };

    auto packW = [&](const float* w, int cout, int cin, int kt, int mode) -> _Float16* {
        int coutP = rup32(cout), cinP = rup32(cin);
        size_t tot = (size_t)kt * coutP * cinP;
        _Float16* buf = alloc16(tot);
        pack_w_kernel<<<dim3((unsigned)((tot + 255) / 256)), 256, 0, stream>>>(
            w, buf, cout, cin, kt, coutP, cinP, mode);
        return buf;
    };

    auto gemm = [&](const _Float16* Wp, const _Float16* X, _Float16* Y,
                    const _Float16* Res, const float* bias, const float* gamma,
                    const float* beta, float* o32, int o32s,
                    int Tin, int Tout, int Vv, int CinP, int CoutP, int cout,
                    int taps, int pad, int relu) {
        int nmT = CoutP / 16;                        // always even (CoutP % 32 == 0)
        int mblk = (nmT % 4 == 0) ? 4 : 2;
        long tiles = (long)((N * Tout * Vv + 15) / 16) * (nmT / mblk);
        unsigned blocks = (unsigned)((tiles + 7) / 8);
        if (mblk == 4)
            gemm_wmma_kernel<4><<<dim3(blocks), 256, 0, stream>>>(
                Wp, X, Y, Res, bias, gamma, beta, o32, o32s,
                N, Tin, Tout, Vv, CinP, CoutP, cout, taps, pad, relu);
        else
            gemm_wmma_kernel<2><<<dim3(blocks), 256, 0, stream>>>(
                Wp, X, Y, Res, bias, gamma, beta, o32, o32s,
                N, Tin, Tout, Vv, CinP, CoutP, cout, taps, pad, relu);
    };

    // --- persistent activation scratch (ping-pong) -------------------------
    const size_t NTV = (size_t)N * T * V;                  // 92160 columns
    _Float16* actA = alloc16(NTV * 256);
    _Float16* actB = alloc16(NTV * 256);
    _Float16* Ybuf = alloc16(NTV * 768);                   // gcn intermediate
    _Float16* Zbuf = alloc16(NTV * 256);
    _Float16* Rbuf = alloc16(NTV * 256);

    auto st_block = [&](const _Float16* Xact, _Float16* Out, const float* A,
                        STIdx ix, int cin, int cout, bool has_res) {
        int CinP = rup32(cin), CoutP = rup32(cout), KCP = rup32(K * cout);
        if (has_res) {
            _Float16* Wr = packW(F(ix.rw), cout, cin, 1, 0);
            gemm(Wr, Xact, Rbuf, nullptr, F(ix.rb), F(ix.rg), F(ix.rbb), nullptr, 0,
                 T, T, V, CinP, CoutP, cout, 1, 0, 0);
        }
        _Float16* Wg = packW(F(ix.gw), K * cout, cin, 1, 0);
        gemm(Wg, Xact, Ybuf, nullptr, F(ix.gb), nullptr, nullptr, nullptr, 0,
             T, T, V, CinP, KCP, K * cout, 1, 0, 0);
        gcn_einsum_kernel<<<dim3(N * T), 256, 0, stream>>>(
            Ybuf, A, F(ix.imp), F(ix.g1), F(ix.b1), Zbuf, N, T, V, cout, KCP, CoutP);
        _Float16* Wt = packW(F(ix.tw), cout, cout, 9, 0);
        gemm(Wt, Zbuf, Out, has_res ? Rbuf : nullptr, F(ix.tb), F(ix.g2), F(ix.b2),
             nullptr, 0, T, T, V, CoutP, CoutP, cout, 9, 4, 1);
    };

    // --- stage 0: data_bn + layout -----------------------------------------
    {
        size_t tot = NTV * 32;
        prep_pose_kernel<<<dim3((unsigned)((tot + 255) / 256)), 256, 0, stream>>>(
            F(1), F(5), F(6), actA, N, T, V, P, 32);
    }

    // --- encoder / reconstructor blocks ------------------------------------
    STIdx e0{ 7,  8,  9, 10, 11, 12, 13, -1, -1, -1, -1, 14, 15};
    STIdx e1{16, 17, 18, 19, 20, 21, 22, 23, 24, 25, 26, 27, 28};
    STIdx e2{29, 30, 31, 32, 33, 34, 35, 36, 37, 38, 39, 40, 41};
    STIdx r0{42, 43, 44, 45, 46, 47, 48, -1, -1, -1, -1, 49, 50};
    STIdx r1{51, 52, 53, 54, 55, 56, 57, 58, 59, 60, 61, 62, 63};
    STIdx r2{64, 65, 66, 67, 68, 69, 70, 71, 72, 73, 74, 75, 76};
    STIdx r3{77, 78, 79, 80, 81, 82, 83, 84, 85, 86, 87, 88, 89};

    st_block(actA, actB, F(3), e0,   3,  64, false);
    st_block(actB, actA, F(3), e1,  64, 128, true);
    st_block(actA, actB, F(3), e2, 128, 256, true);
    st_block(actB, actA, F(4), r0, 256, 128, false);
    st_block(actA, actB, F(4), r1, 128,  64, true);
    st_block(actB, actA, F(4), r2,  64,  32, true);
    st_block(actA, actB, F(4), r3,  32,   3, true);       // x2 in actB (CP=32)

    // --- prediction head ----------------------------------------------------
    const int cols10 = N * 10;
    _Float16* locX = alloc16((size_t)cols10 * 32);
    pack_cols_kernel<<<dim3((unsigned)((cols10 * 32 + 255) / 256)), 256, 0, stream>>>(
        F(0), locX, cols10, L, 32);
    _Float16* gfX = alloc16((size_t)cols10 * 32);
    pack_cols_kernel<<<dim3((unsigned)((cols10 * 32 + 255) / 256)), 256, 0, stream>>>(
        F(2), gfX, cols10, G, 32);
    _Float16* poseX = alloc16((size_t)cols10 * 64);
    pose_in_kernel<<<dim3((unsigned)((cols10 * 64 + 255) / 256)), 256, 0, stream>>>(
        actB, poseX, N, T, V, P, 32, 64);

    // cbr: conv(k) + bias + bn + relu ; weight dict leaves at wi: b,bb,g,w
    auto cbr = [&](const _Float16* X, int Tin, int Tout, int cin, int cout, int wi,
                   int taps, int pad, float* o32, int o32s) -> _Float16* {
        int CinP = rup32(cin), CoutP = rup32(cout);
        _Float16* Wc = packW(F(wi + 3), cout, cin, taps, 0);
        _Float16* Yb = alloc16((size_t)N * Tout * CoutP);
        gemm(Wc, X, Yb, nullptr, F(wi + 0), F(wi + 2), F(wi + 1), o32, o32s,
             Tin, Tout, 1, CinP, CoutP, cout, taps, pad, 1);
        return Yb;
    };
    // dbr: ConvTranspose(k=3) == conv(flipped, pad=2), Tout = Tin + 2
    auto dbr = [&](const _Float16* X, int Tin, int cin, int cout, int wi) -> _Float16* {
        int CinP = rup32(cin), CoutP = rup32(cout);
        _Float16* Wc = packW(F(wi + 3), cout, cin, 3, 1);
        _Float16* Yb = alloc16((size_t)N * (Tin + 2) * CoutP);
        gemm(Wc, X, Yb, nullptr, F(wi + 0), F(wi + 2), F(wi + 1), nullptr, 0,
             Tin, Tin + 2, 1, CinP, CoutP, cout, 3, 2, 1);
        return Yb;
    };

    const int locI[4]  = {138, 142, 146, 150};
    const int poseI[4] = {154, 158, 162, 166};
    const int gfI[4]   = {118, 122, 126, 130};
    auto stack4 = [&](_Float16* X0, int cin0, const int* wi) -> _Float16* {
        int chs[5] = {cin0, 32, 64, 128, 128};
        _Float16* x = X0;
        int Tc = 10;
        for (int i = 0; i < 4; ++i) {
            x = cbr(x, Tc, Tc - 2, chs[i], chs[i + 1], wi[i], 3, 0, nullptr, 0);
            Tc -= 2;
        }
        return x;
    };
    _Float16* a  = stack4(locX,  2, locI);
    _Float16* pz = stack4(poseX, 54, poseI);
    _Float16* c  = stack4(gfX,  24, gfI);

    _Float16* fcat = alloc16((size_t)N * 2 * 384);
    concat3_kernel<<<dim3((unsigned)((N * 2 * 384 + 255) / 256)), 256, 0, stream>>>(
        a, pz, c, fcat, N * 2);

    _Float16* f1 = cbr(fcat, 2, 2, 384, 128, 134, 1, 0, nullptr, 0);   // inter
    _Float16* f2 = cbr(f1,   2, 2, 128, 256,  90, 1, 0, nullptr, 0);   // dec_c0
    _Float16* f3 = cbr(f2,   2, 2, 256, 256,  94, 1, 0, nullptr, 0);   // dec_c1
    _Float16* g0 = dbr(f3,  2, 256, 256,  98);                         // T 2->4
    _Float16* g1 = dbr(g0,  4, 256, 128, 102);                         // 4->6
    _Float16* g2 = dbr(g1,  6, 128,  64, 106);                         // 6->8
    _Float16* g3 = dbr(g2,  8,  64,  32, 110);                         // 8->10
    // dec_out (1x1, cout=2) writes fp32 directly into d_out as (N, 10, 2)
    cbr(g3, 10, 10, 32, 2, 114, 1, 0, (float*)d_out, 2);
}
